// EfficientAttentionV2_29480655519764
// MI455X (gfx1250) — compile-verified, matrix operand order checked by reading, not run on hardware
//
#include <hip/hip_runtime.h>

// ---------------- problem constants (from reference) ----------------
#define N_TOK   32768      // sum(COUNTS)
#define C_DIM   768
#define THREE_C 2304
#define H_HEADS 16
#define D_HEAD  48
#define DP      64         // d padded to 64 for K=32 WMMA chunks
#define K_PART  512
#define NP_PAD  33792      // padded rows (66 partitions * 512)
#define PH_CNT  1056       // 66 partitions * 16 heads

typedef __attribute__((ext_vector_type(16))) __bf16 v16bf;
typedef __attribute__((ext_vector_type(8)))  float  v8f;
typedef unsigned int v4ui __attribute__((ext_vector_type(4)));
typedef int          v4i  __attribute__((ext_vector_type(4)));
typedef int          v8i  __attribute__((ext_vector_type(8)));

union ABu { uint4 q[2]; v16bf v; unsigned short s[16]; };
static_assert(sizeof(ABu) == 32, "ABu size");

__device__ __forceinline__ unsigned short f2bf(float f) {
  unsigned u = __float_as_uint(f);
  u += 0x7FFFu + ((u >> 16) & 1u);     // round-to-nearest-even
  return (unsigned short)(u >> 16);
}

__device__ __forceinline__ v8f wmma_bf16(const ABu& a, const ABu& b, v8f c) {
  return __builtin_amdgcn_wmma_f32_16x16x32_bf16(false, a.v, false, b.v,
                                                 (short)0, c, false, false);
}

// ---------------- Tensor Data Mover (async 1D tile -> LDS) ----------------
#if defined(__AMDGCN__) && __has_builtin(__builtin_amdgcn_tensor_load_to_lds)
#define USE_TDM 1
#else
#define USE_TDM 0
#endif

#if USE_TDM
// D# per CDNA5 ISA: group0 = {count/lds_addr/global_addr/type},
// group1 = {data_size, tensor/tile dims, strides}. 1D copy of nelem 2-byte elements
// expressed as a 2D tile [nelem x 1].
__device__ __forceinline__ void tdm_load_to_lds_1d(unsigned lds_byte_off,
                                                   const void* gsrc,
                                                   unsigned nelem) {
  unsigned long long ga = (unsigned long long)(size_t)gsrc;
  v4ui g0;
  g0[0] = 1u;                                             // count=1 (user D#)
  g0[1] = lds_byte_off;                                   // lds_addr (bytes)
  g0[2] = (unsigned)(ga & 0xffffffffu);                   // global_addr[31:0]
  g0[3] = (unsigned)((ga >> 32) & 0x01ffffffu) | (2u << 30); // addr[56:32] | type=2
  v8i g1;
  g1[0] = (int)(1u << 16);                                // data_size=1 (2 bytes)
  g1[1] = (int)((nelem & 0xffffu) << 16);                 // tensor_dim0[15:0]
  g1[2] = (int)((nelem >> 16) | (1u << 16));              // tensor_dim0[31:16] | tensor_dim1[15:0]=1
  g1[3] = (int)((nelem & 0xffffu) << 16);                 // tile_dim0 = nelem
  g1[4] = 1;                                              // tile_dim1 = 1
  g1[5] = (int)nelem;                                     // tensor_dim0_stride[31:0]
  g1[6] = 0;
  g1[7] = 0;
  v4i z4 = {0, 0, 0, 0};
#if __clang_major__ >= 23
  v8i z8 = {0, 0, 0, 0, 0, 0, 0, 0};
  __builtin_amdgcn_tensor_load_to_lds(g0, g1, z4, z4, z8, 0);
#else
  __builtin_amdgcn_tensor_load_to_lds(g0, g1, z4, z4, 0);
#endif
}

__device__ __forceinline__ void tdm_wait0() {
#if __has_builtin(__builtin_amdgcn_s_wait_tensorcnt)
  __builtin_amdgcn_s_wait_tensorcnt(0);
#else
  asm volatile("s_wait_tensorcnt 0x0" ::: "memory");
#endif
}
#endif // USE_TDM

// ---------------- workspace layout (bytes) ----------------
static constexpr size_t OFF_XB   = 0;                                      // x in bf16
static constexpr size_t OFF_WQT  = OFF_XB  + (size_t)N_TOK * C_DIM * 2;    // w_qkv^T bf16 [3C][C]
static constexpr size_t OFF_WPT  = OFF_WQT + (size_t)THREE_C * C_DIM * 2;  // w_proj^T bf16 [C][C]
static constexpr size_t OFF_Q    = OFF_WPT + (size_t)C_DIM * C_DIM * 2;    // q  [ph][512][64]
static constexpr size_t OFF_K    = OFF_Q   + (size_t)PH_CNT * K_PART * DP * 2; // k [ph][512][64]
static constexpr size_t OFF_VT   = OFF_K   + (size_t)PH_CNT * K_PART * DP * 2; // v^T [ph][64][512]
static constexpr size_t OFF_FEAT = OFF_VT  + (size_t)PH_CNT * K_PART * DP * 2; // feat bf16 [NP][C]
// total ~315 MB of workspace

// ---------------- K1a: f32 -> bf16 ----------------
__global__ __launch_bounds__(256) void k_convert(const float* __restrict__ src,
                                                 unsigned short* __restrict__ dst,
                                                 int n) {
  for (int i = blockIdx.x * 256 + threadIdx.x; i < n; i += gridDim.x * 256)
    dst[i] = f2bf(src[i]);
}

// ---------------- K1b: transpose + convert: dst[n][k] = src[k][n] ----------------
__global__ __launch_bounds__(256) void k_transpose(const float* __restrict__ src,
                                                   unsigned short* __restrict__ dst,
                                                   int kdim, int ndim) {
  const int total = kdim * ndim;
  for (int i = blockIdx.x * 256 + threadIdx.x; i < total; i += gridDim.x * 256) {
    int k = i / ndim, n = i - k * ndim;
    dst[(size_t)n * kdim + k] = f2bf(src[i]);
  }
}

// ---------------- K1c: zero the d=48..63 pad of q/k buffers ----------------
__global__ __launch_bounds__(256) void k_zero_pad(unsigned short* __restrict__ qb,
                                                  unsigned short* __restrict__ kb) {
  const int rows  = PH_CNT * K_PART;   // 540672 rows per buffer
  const int total = rows * 2 * 2;      // 2 uint4 per row pad, 2 buffers
  uint4 z; z.x = z.y = z.z = z.w = 0u;
  for (int i = blockIdx.x * 256 + threadIdx.x; i < total; i += gridDim.x * 256) {
    int buf = i / (rows * 2);
    int rem = i - buf * rows * 2;
    int r = rem >> 1, wch = rem & 1;
    unsigned short* base = buf ? kb : qb;
    *(uint4*)(base + (size_t)r * DP + D_HEAD + wch * 8) = z;
  }
}

// ---------------- K2: QKV GEMM (+order gather) + bias + RoPE + scatter ----------------
// One wave computes one 16x16 output tile of qkv[NP_PAD][2304].
// Each 16-col tile lies fully inside one (t, head, 16-d rope part) since 48 = 3*16.
__global__ __launch_bounds__(256) void k_qkv_rope(
    const unsigned short* __restrict__ xb,
    const unsigned short* __restrict__ wqT,
    const float* __restrict__ bqkv,
    const int* __restrict__ order,
    const int* __restrict__ pos,
    unsigned short* __restrict__ qbuf,
    unsigned short* __restrict__ kbuf,
    unsigned short* __restrict__ vtbuf) {
  const int wave = threadIdx.x >> 5;
  const int lane = threadIdx.x & 31;
  const int half = lane >> 4;
  const int l16  = lane & 15;
  const int off8 = half << 3;

  const int rt = blockIdx.x;               // row tile  [0, 2112)
  const int ct = blockIdx.y * 8 + wave;    // col tile  [0, 144)
  const int rbase = rt << 4;
  const int j0    = ct << 4;

  const int r   = rbase + l16;
  const int tok = order[r];                // gather fused into A fetch
  const unsigned short* arow = xb  + (size_t)tok * C_DIM;
  const unsigned short* brow = wqT + (size_t)(j0 + l16) * C_DIM;

  v8f acc{};
  #pragma unroll 4
  for (int kk = 0; kk < C_DIM; kk += 32) {
    __builtin_prefetch(arow + kk + 128, 0, 1);
    __builtin_prefetch(brow + kk + 128, 0, 1);
    ABu a, b;
    a.q[0] = *(const uint4*)(arow + kk + off8);
    a.q[1] = *(const uint4*)(arow + kk + 16 + off8);
    b.q[0] = *(const uint4*)(brow + kk + off8);
    b.q[1] = *(const uint4*)(brow + kk + 16 + off8);
    acc = wmma_bf16(a, b, acc);
  }

  const float bias = bqkv[j0 + l16];
  float vals[8];
  #pragma unroll
  for (int j = 0; j < 8; ++j) vals[j] = acc[j] + bias;

  const int t     = j0 / C_DIM;            // 0=q 1=k 2=v
  const int rem   = j0 - t * C_DIM;
  const int h     = rem / D_HEAD;
  const int dbase = rem - h * D_HEAD;      // 0, 16, 32
  const int part  = rbase >> 9;
  const int ph    = (part << 4) + h;
  const int rloc0 = (rbase & (K_PART - 1)) + (half << 3);

  if (t < 2) {
    // RoPE on this 16-wide part; rotate-half partner is lane^8 in C layout
    const int i3 = dbase >> 4;             // pos component for this part
    const float invf = __powf(100.0f, -(float)(l16 & 7) * 0.125f);
    #pragma unroll
    for (int j = 0; j < 8; ++j) {
      const int rr = rbase + j + (half << 3);
      const float ang = (float)pos[rr * 3 + i3] * invf;
      float sn, cs;
      __sincosf(ang, &sn, &cs);
      const float partner = __shfl_xor(vals[j], 8, 32);
      const float rot = (l16 < 8) ? -partner : partner;
      vals[j] = vals[j] * cs + rot * sn;
    }
    if (t == 0) {
      #pragma unroll
      for (int j = 0; j < 8; ++j) vals[j] *= 0.14433756729740643f; // 1/sqrt(48)
    }
    unsigned short* dst = (t == 0) ? qbuf : kbuf;
    #pragma unroll
    for (int j = 0; j < 8; ++j)
      dst[((size_t)ph * K_PART + rloc0 + j) * DP + dbase + l16] = f2bf(vals[j]);
  } else {
    // v stored transposed [64][512]: per-lane rows are contiguous -> one b128 store
    uint4 u;
    u.x = (unsigned)f2bf(vals[0]) | ((unsigned)f2bf(vals[1]) << 16);
    u.y = (unsigned)f2bf(vals[2]) | ((unsigned)f2bf(vals[3]) << 16);
    u.z = (unsigned)f2bf(vals[4]) | ((unsigned)f2bf(vals[5]) << 16);
    u.w = (unsigned)f2bf(vals[6]) | ((unsigned)f2bf(vals[7]) << 16);
    *(uint4*)(vtbuf + ((size_t)ph * DP + dbase + l16) * K_PART + rloc0) = u;
  }
}

// ---------------- K3: per-(partition,head) flash attention ----------------
// S^T = k . q^T so that the post-softmax C-layout doubles as the A operand of P.V.
__global__ __launch_bounds__(256) void k_attention(
    const unsigned short* __restrict__ qbuf,
    const unsigned short* __restrict__ kbuf,
    const unsigned short* __restrict__ vtbuf,
    unsigned short* __restrict__ featb) {
  __shared__ unsigned short shK[K_PART * DP];   // 64 KB
  __shared__ unsigned short shV[DP * K_PART];   // 64 KB (v transposed)

  const int ph   = blockIdx.x;
  const int part = ph >> 4;
  const int h    = ph & 15;

#if USE_TDM
  // Tensor Data Mover: two descriptor-driven DMAs fill k and v^T slices.
  // EXEC is ignored by TDM; issue once from wave 0, wait on TENSORcnt there,
  // then barrier so every wave sees the LDS tiles.
  if (threadIdx.x < 32) {
    tdm_load_to_lds_1d((unsigned)(size_t)(void*)shK,
                       kbuf + (size_t)ph * K_PART * DP, K_PART * DP);
    tdm_load_to_lds_1d((unsigned)(size_t)(void*)shV,
                       vtbuf + (size_t)ph * DP * K_PART, DP * K_PART);
    tdm_wait0();
  }
#else
  {
    const uint4* gk = (const uint4*)(kbuf  + (size_t)ph * K_PART * DP);
    const uint4* gv = (const uint4*)(vtbuf + (size_t)ph * DP * K_PART);
    uint4* sk = (uint4*)shK;
    uint4* sv = (uint4*)shV;
    for (int i = threadIdx.x; i < (K_PART * DP) / 8; i += 256) {
      sk[i] = gk[i];
      sv[i] = gv[i];
    }
  }
#endif
  __syncthreads();

  const int wave = threadIdx.x >> 5;
  const int lane = threadIdx.x & 31;
  const int half = lane >> 4;
  const int l16  = lane & 15;
  const int off8 = half << 3;

  for (int qb = wave; qb < K_PART / 16; qb += 8) {
    const int qbase = qb << 4;
    // q B-operand (B[d][q], q stored [512][64]) held in registers for whole loop
    ABu qB0, qB1;
    const unsigned short* qrow = qbuf + ((size_t)ph * K_PART + qbase + l16) * DP;
    qB0.q[0] = *(const uint4*)(qrow + off8);
    qB0.q[1] = *(const uint4*)(qrow + 16 + off8);
    qB1.q[0] = *(const uint4*)(qrow + 32 + off8);
    qB1.q[1] = *(const uint4*)(qrow + 48 + off8);

    v8f O0{}, O1{}, O2{};
    float mrun = -1e30f, lrun = 0.0f;

    for (int kt = 0; kt < K_PART; kt += 32) {
      v8f s0{}, s1{};
      {
        const unsigned short* krow = shK + (size_t)(kt + l16) * DP;
        ABu kA;
        kA.q[0] = *(const uint4*)(krow + off8);
        kA.q[1] = *(const uint4*)(krow + 16 + off8);
        s0 = wmma_bf16(kA, qB0, s0);
        kA.q[0] = *(const uint4*)(krow + 32 + off8);
        kA.q[1] = *(const uint4*)(krow + 48 + off8);
        s0 = wmma_bf16(kA, qB1, s0);
      }
      {
        const unsigned short* krow = shK + (size_t)(kt + 16 + l16) * DP;
        ABu kA;
        kA.q[0] = *(const uint4*)(krow + off8);
        kA.q[1] = *(const uint4*)(krow + 16 + off8);
        s1 = wmma_bf16(kA, qB0, s1);
        kA.q[0] = *(const uint4*)(krow + 32 + off8);
        kA.q[1] = *(const uint4*)(krow + 48 + off8);
        s1 = wmma_bf16(kA, qB1, s1);
      }

      // online softmax over keys; per-q state lives in lanes l and l+16
      float lm = -1e30f;
      #pragma unroll
      for (int j = 0; j < 8; ++j) lm = fmaxf(lm, fmaxf(s0[j], s1[j]));
      lm = fmaxf(lm, __shfl_xor(lm, 16, 32));
      const float mnew  = fmaxf(mrun, lm);
      const float alpha = __expf(mrun - mnew);
      mrun = mnew;

      float p0[8], p1[8], psum = 0.0f;
      #pragma unroll
      for (int j = 0; j < 8; ++j) {
        p0[j] = __expf(s0[j] - mnew);
        p1[j] = __expf(s1[j] - mnew);
        psum += p0[j] + p1[j];
      }
      lrun = lrun * alpha + psum;

      // broadcast alpha per accumulator row (O row q = j + half*8 lives at lane q)
      float aj[8];
      #pragma unroll
      for (int j = 0; j < 8; ++j) aj[j] = __shfl(alpha, j + off8, 32);
      #pragma unroll
      for (int j = 0; j < 8; ++j) { O0[j] *= aj[j]; O1[j] *= aj[j]; O2[j] *= aj[j]; }

      // two consecutive S^T C-tiles concatenate into P's 16x32 A fragment
      ABu pa;
      #pragma unroll
      for (int e = 0; e < 8; ++e) { pa.s[e] = f2bf(p0[e]); pa.s[e + 8] = f2bf(p1[e]); }

      {
        const unsigned short* vrow = shV + (size_t)l16 * K_PART + kt;
        ABu vB;
        vB.q[0] = *(const uint4*)(vrow + off8);
        vB.q[1] = *(const uint4*)(vrow + 16 + off8);
        O0 = wmma_bf16(pa, vB, O0);
        vrow = shV + (size_t)(16 + l16) * K_PART + kt;
        vB.q[0] = *(const uint4*)(vrow + off8);
        vB.q[1] = *(const uint4*)(vrow + 16 + off8);
        O1 = wmma_bf16(pa, vB, O1);
        vrow = shV + (size_t)(32 + l16) * K_PART + kt;
        vB.q[0] = *(const uint4*)(vrow + off8);
        vB.q[1] = *(const uint4*)(vrow + 16 + off8);
        O2 = wmma_bf16(pa, vB, O2);
      }
    }

    const float ltot = lrun + __shfl_xor(lrun, 16, 32);
    const float linv = 1.0f / ltot;
    float rj[8];
    #pragma unroll
    for (int j = 0; j < 8; ++j) rj[j] = __shfl(linv, j + off8, 32);

    const size_t rowbase = (size_t)part * K_PART + qbase + (half << 3);
    #pragma unroll
    for (int j = 0; j < 8; ++j) {
      unsigned short* orow = featb + (rowbase + j) * C_DIM + h * D_HEAD;
      orow[l16]      = f2bf(O0[j] * rj[j]);
      orow[16 + l16] = f2bf(O1[j] * rj[j]);
      orow[32 + l16] = f2bf(O2[j] * rj[j]);
    }
  }
}

// ---------------- K4: output projection (+inverse un-permute) ----------------
__global__ __launch_bounds__(256) void k_proj(
    const unsigned short* __restrict__ featb,
    const unsigned short* __restrict__ wpT,
    const float* __restrict__ bproj,
    const int* __restrict__ inverse,
    float* __restrict__ out) {
  const int wave = threadIdx.x >> 5;
  const int lane = threadIdx.x & 31;
  const int half = lane >> 4;
  const int l16  = lane & 15;
  const int off8 = half << 3;

  const int rt = blockIdx.x;
  const int ct = blockIdx.y * 8 + wave;
  const int r  = (rt << 4) + l16;
  const int padrow = inverse[r];
  const unsigned short* arow = featb + (size_t)padrow * C_DIM;
  const unsigned short* brow = wpT + (size_t)((ct << 4) + l16) * C_DIM;

  v8f acc{};
  #pragma unroll 4
  for (int kk = 0; kk < C_DIM; kk += 32) {
    __builtin_prefetch(arow + kk + 128, 0, 1);
    __builtin_prefetch(brow + kk + 128, 0, 1);
    ABu a, b;
    a.q[0] = *(const uint4*)(arow + kk + off8);
    a.q[1] = *(const uint4*)(arow + kk + 16 + off8);
    b.q[0] = *(const uint4*)(brow + kk + off8);
    b.q[1] = *(const uint4*)(brow + kk + 16 + off8);
    acc = wmma_bf16(a, b, acc);
  }

  const float bias = bproj[(ct << 4) + l16];
  const int rowb = (rt << 4) + (half << 3);
  #pragma unroll
  for (int j = 0; j < 8; ++j)
    out[(size_t)(rowb + j) * C_DIM + (ct << 4) + l16] = acc[j] + bias;
}

// ---------------- host launcher ----------------
extern "C" void kernel_launch(void* const* d_in, const int* in_sizes, int n_in,
                              void* d_out, int out_size, void* d_ws, size_t ws_size,
                              hipStream_t stream) {
  const float* x      = (const float*)d_in[0];
  const float* w_qkv  = (const float*)d_in[1];
  const float* b_qkv  = (const float*)d_in[2];
  const float* w_proj = (const float*)d_in[3];
  const float* b_proj = (const float*)d_in[4];
  const int*   order  = (const int*)d_in[5];
  const int*   inv    = (const int*)d_in[6];
  const int*   pos    = (const int*)d_in[7];
  float* out = (float*)d_out;

  char* ws = (char*)d_ws;
  unsigned short* xb    = (unsigned short*)(ws + OFF_XB);
  unsigned short* wqT   = (unsigned short*)(ws + OFF_WQT);
  unsigned short* wpT   = (unsigned short*)(ws + OFF_WPT);
  unsigned short* qbuf  = (unsigned short*)(ws + OFF_Q);
  unsigned short* kbuf  = (unsigned short*)(ws + OFF_K);
  unsigned short* vtbuf = (unsigned short*)(ws + OFF_VT);
  unsigned short* featb = (unsigned short*)(ws + OFF_FEAT);

  k_convert  <<<dim3(2048), dim3(256), 0, stream>>>(x, xb, N_TOK * C_DIM);
  k_transpose<<<dim3(1024), dim3(256), 0, stream>>>(w_qkv, wqT, C_DIM, THREE_C);
  k_transpose<<<dim3(512),  dim3(256), 0, stream>>>(w_proj, wpT, C_DIM, C_DIM);
  k_zero_pad <<<dim3(1024), dim3(256), 0, stream>>>(qbuf, kbuf);

  k_qkv_rope <<<dim3(NP_PAD / 16, (THREE_C / 16) / 8), dim3(256), 0, stream>>>(
      xb, wqT, b_qkv, order, pos, qbuf, kbuf, vtbuf);

  k_attention<<<dim3(PH_CNT), dim3(256), 0, stream>>>(qbuf, kbuf, vtbuf, featb);

  k_proj     <<<dim3(N_TOK / 16, (C_DIM / 16) / 8), dim3(256), 0, stream>>>(
      featb, wpT, b_proj, inv, out);

  (void)in_sizes; (void)n_in; (void)out_size; (void)ws_size;
}
